// TBCNNCell_85246510891461
// MI455X (gfx1250) — matrix operand order repaired and checked by reading, not compiled, not run on hardware
//
#include <hip/hip_runtime.h>
#include <hip/hip_bf16.h>

#define N_NODES 131072
#define HSZ 256
#define MB 64  // rows per GEMM block

#define SB_ELEMS (3 * 16 * 512)   // 24576 bf16 per buffer (48 KB)
#define SA_ELEMS (MB * 32)        //  2048 bf16 per buffer (4 KB)
#define SMEM_BYTES ((2 * SB_ELEMS + 2 * SA_ELEMS) * 2)   // 106496 B dynamic LDS

#if __has_builtin(__builtin_amdgcn_global_load_async_to_lds_b128)
#define HAVE_ASYNC_LDS 1
#endif

typedef __attribute__((ext_vector_type(16))) __bf16 v16bf;
typedef __attribute__((ext_vector_type(8)))  __bf16 v8bf;
typedef __attribute__((ext_vector_type(2)))  __bf16 v2bf;
typedef __attribute__((ext_vector_type(8)))  float  v8f;
typedef __attribute__((ext_vector_type(4)))  int    v4i;

__device__ __forceinline__ unsigned short f2bf_rne(float f) {
    union { float f; unsigned u; } v; v.f = f;
    unsigned u = v.u;
    u += 0x7fffu + ((u >> 16) & 1u);
    return (unsigned short)(u >> 16);
}

// pack two floats -> one dword of bf16 (a low, b high)
__device__ __forceinline__ unsigned pack_bf16(float a, float b) {
#if __has_builtin(__builtin_amdgcn_cvt_pk_bf16_f32)
    union { v2bf v; unsigned u; } r;
    r.v = __builtin_amdgcn_cvt_pk_bf16_f32(a, b);
    return r.u;
#else
    return (unsigned)f2bf_rne(a) | ((unsigned)f2bf_rne(b) << 16);
#endif
}

#ifdef HAVE_ASYNC_LDS
typedef __attribute__((address_space(1))) v4i* gv4i_p;   // global (AS1)
typedef __attribute__((address_space(3))) v4i* lv4i_p;   // LDS (AS3)
// per-lane 16B async DMA copy global -> LDS (tracked by ASYNCcnt)
__device__ __forceinline__ void async_copy16(const void* g, void* l) {
    void* gg = const_cast<void*>(g);
    __builtin_amdgcn_global_load_async_to_lds_b128((gv4i_p)gg, (lv4i_p)l, 0, 0);
}
#endif

// ---------------- zero / relu elementwise ----------------
__global__ void zero_f4(float4* __restrict__ p, int n4) {
    int i = blockIdx.x * blockDim.x + threadIdx.x;
    if (i < n4) p[i] = make_float4(0.f, 0.f, 0.f, 0.f);
}

__global__ void relu_f4(float4* __restrict__ p, int n4) {
    int i = blockIdx.x * blockDim.x + threadIdx.x;
    if (i < n4) {
        float4 v = p[i];
        v.x = fmaxf(v.x, 0.f); v.y = fmaxf(v.y, 0.f);
        v.z = fmaxf(v.z, 0.f); v.w = fmaxf(v.w, 0.f);
        p[i] = v;
    }
}

// ---------------- weight pack: f32 [K,N] -> bf16 B-fragment layout ----------------
__global__ void pack_weights(const float* __restrict__ Wl,
                             const float* __restrict__ Wr,
                             const float* __restrict__ Wt,
                             unsigned short* __restrict__ wsW) {
    int t = blockIdx.x * blockDim.x + threadIdx.x;
    if (t >= 3 * 256 * 256) return;
    int mat = t >> 16;
    int rem = t & 0xffff;          // k*256 + n
    int k = rem >> 8, n = rem & 255;
    const float* W = (mat == 0) ? Wl : (mat == 1 ? Wr : Wt);
    float v = W[rem];
    int s = k >> 5, kl = k & 31;
    int ct = n >> 4, nl = n & 15;
    int half = kl >> 4, j = kl & 15;
    int lane = half * 16 + nl;
    int dst = ((mat * 8 + s) * 16 + ct) * 512 + lane * 16 + j;
    wsW[dst] = f2bf_rne(v);
}

// ---------------- fused 3-GEMM + edge message scatter ----------------
__global__ __launch_bounds__(512)
__attribute__((amdgpu_waves_per_eu(4, 4)))
void gemm_tbcnn(
    const float* __restrict__ hmat,
    const uint4* __restrict__ W4,           // packed bf16 weights as uint4
    const float* __restrict__ bias,
    const int* __restrict__ edge_dest,
    const int* __restrict__ child_pos,
    const int* __restrict__ child_cnt,
    float* __restrict__ out)
{
    extern __shared__ __bf16 smem[];
    __bf16* sB = smem;                       // 2 buffers of SB_ELEMS
    __bf16* sA = smem + 2 * SB_ELEMS;        // 2 buffers of SA_ELEMS
    uint4*  sB4 = (uint4*)sB;

    const int tid  = threadIdx.x;
    const int lane = tid & 31;
    const int wave = tid >> 5;
    const int rt   = wave >> 2;              // row tile 0..3
    const int cq   = wave & 3;               // column quarter 0..3
    const int rowbase = blockIdx.x * MB;
    const int m16  = lane & 15;
    const int half = lane >> 4;

    // A staging: thread loads one float4: row = tid>>3, kk = (tid&7)*4
    const int arow = tid >> 3;
    const int akk  = (tid & 7) * 4;
    const float* aPtrBase = hmat + (size_t)(rowbase + arow) * HSZ + akk;

    const int aoff = (rt * 16 + m16) * 32 + half * 16;  // A frag elem offset
    const int boff = (cq * 4) * 512 + lane * 16;        // B frag base elem offset

    v8f acc[3][4];
    #pragma unroll
    for (int m = 0; m < 3; ++m)
        #pragma unroll
        for (int c = 0; c < 4; ++c)
            acc[m][c] = (v8f){0.f,0.f,0.f,0.f,0.f,0.f,0.f,0.f};

    // ---- prologue: stage K-step 0 into LDS buffer 0 ----
    {
#ifdef HAVE_ASYNC_LDS
        #pragma unroll
        for (int mat = 0; mat < 3; ++mat)
            #pragma unroll
            for (int j = 0; j < 2; ++j)
                async_copy16(W4 + tid + mat * 8192 + j * 512,
                             sB4 + tid + mat * 1024 + j * 512);
#else
        uint4 b0[6];
        #pragma unroll
        for (int mat = 0; mat < 3; ++mat)
            #pragma unroll
            for (int j = 0; j < 2; ++j)
                b0[mat * 2 + j] = W4[tid + mat * 8192 + j * 512];
        #pragma unroll
        for (int mat = 0; mat < 3; ++mat)
            #pragma unroll
            for (int j = 0; j < 2; ++j)
                sB4[tid + mat * 1024 + j * 512] = b0[mat * 2 + j];
#endif
        float4 a0 = *(const float4*)aPtrBase;
        uint2 pk;
        pk.x = pack_bf16(a0.x, a0.y);
        pk.y = pack_bf16(a0.z, a0.w);
        *(uint2*)&sA[arow * 32 + akk] = pk;
#ifdef HAVE_ASYNC_LDS
        asm volatile("s_wait_asynccnt 0x0" ::: "memory");
#endif
    }
    __syncthreads();

    #pragma unroll 2
    for (int s = 0; s < 8; ++s) {
        const int cur = s & 1, nxt = cur ^ 1;
        const __bf16* sBc = sB + cur * SB_ELEMS;
        const __bf16* sAc = sA + cur * SA_ELEMS;

        // ---- kick off step s+1 transfers (ride under the WMMAs) ----
#ifndef HAVE_ASYNC_LDS
        uint4 nB[6];
#endif
        float4 nA;
        if (s < 7) {
#ifdef HAVE_ASYNC_LDS
            #pragma unroll
            for (int mat = 0; mat < 3; ++mat)
                #pragma unroll
                for (int j = 0; j < 2; ++j)
                    async_copy16(W4 + tid + (s + 1) * 1024 + mat * 8192 + j * 512,
                                 sB4 + nxt * 3072 + tid + mat * 1024 + j * 512);
#else
            #pragma unroll
            for (int mat = 0; mat < 3; ++mat)
                #pragma unroll
                for (int j = 0; j < 2; ++j)
                    nB[mat * 2 + j] = W4[tid + (s + 1) * 1024 + mat * 8192 + j * 512];
#endif
            nA = *(const float4*)(aPtrBase + (s + 1) * 32);
        }

        // ---- A fragment (contiguous K = half*16 + r, matching packed B) ----
        union { v16bf v; struct { v8bf lo; v8bf hi; } p; } af;
        af.p.lo = *(const v8bf*)(sAc + aoff);
        af.p.hi = *(const v8bf*)(sAc + aoff + 8);

        // ---- 12 WMMAs, hand-pipelined fragment loads (2 reg sets, no copies) ----
        v16bf f0[4], f1[4];
        #pragma unroll
        for (int c = 0; c < 4; ++c)
            f0[c] = *(const v16bf*)&sBc[0 * 16 * 512 + boff + c * 512];
        #pragma unroll
        for (int c = 0; c < 4; ++c)
            f1[c] = *(const v16bf*)&sBc[1 * 16 * 512 + boff + c * 512];
        #pragma unroll
        for (int c = 0; c < 4; ++c)
            acc[0][c] = __builtin_amdgcn_wmma_f32_16x16x32_bf16(
                false, af.v, false, f0[c], (short)0, acc[0][c], false, false);
        #pragma unroll
        for (int c = 0; c < 4; ++c)
            f0[c] = *(const v16bf*)&sBc[2 * 16 * 512 + boff + c * 512];
        #pragma unroll
        for (int c = 0; c < 4; ++c)
            acc[1][c] = __builtin_amdgcn_wmma_f32_16x16x32_bf16(
                false, af.v, false, f1[c], (short)0, acc[1][c], false, false);
        #pragma unroll
        for (int c = 0; c < 4; ++c)
            acc[2][c] = __builtin_amdgcn_wmma_f32_16x16x32_bf16(
                false, af.v, false, f0[c], (short)0, acc[2][c], false, false);

        // ---- commit step s+1 A tile (+ B on the fallback path) ----
        if (s < 7) {
#ifndef HAVE_ASYNC_LDS
            #pragma unroll
            for (int mat = 0; mat < 3; ++mat)
                #pragma unroll
                for (int j = 0; j < 2; ++j)
                    sB4[nxt * 3072 + tid + mat * 1024 + j * 512] = nB[mat * 2 + j];
#endif
            uint2 pk;
            pk.x = pack_bf16(nA.x, nA.y);
            pk.y = pack_bf16(nA.z, nA.w);
            *(uint2*)&sA[nxt * SA_ELEMS + arow * 32 + akk] = pk;
        }
#ifdef HAVE_ASYNC_LDS
        asm volatile("s_wait_asynccnt 0x0" ::: "memory");
#endif
        __syncthreads();
    }

    // ---- epilogue: C/D layout row M = v + 8*half, col N = lane&15 ----
    const int rowt = rowbase + rt * 16;

    float bcol[4];
    #pragma unroll
    for (int c = 0; c < 4; ++c) bcol[c] = bias[cq * 64 + c * 16 + m16];

    #pragma unroll
    for (int v = 0; v < 8; ++v) {
        int r = rowt + v + half * 8;
        float lw = 0.f, rw = 0.f;
        int dest = 0;
        bool has_edge = (r >= 1);
        if (has_edge) {
            int e = r - 1;                 // node r is the source of edge e
            int cnt = child_cnt[e];
            int pos = child_pos[e];
            dest = edge_dest[e];
            if (cnt == 1) { lw = 1.f; rw = 0.f; }
            else {
                float denom = (float)(cnt - 1);
                rw = (float)pos / denom;
                lw = (float)(cnt - 1 - pos) / denom;
            }
        }
        #pragma unroll
        for (int c = 0; c < 4; ++c) {
            int col = cq * 64 + c * 16 + m16;
            float ht = acc[2][c][v] + bcol[c];         // h@W_top + b (added once)
            __hip_atomic_fetch_add(&out[(size_t)r * HSZ + col], ht,
                                   __ATOMIC_RELAXED, __HIP_MEMORY_SCOPE_AGENT);
            if (has_edge) {
                float msg = lw * acc[0][c][v] + rw * acc[1][c][v];
                __hip_atomic_fetch_add(&out[(size_t)dest * HSZ + col], msg,
                                       __ATOMIC_RELAXED, __HIP_MEMORY_SCOPE_AGENT);
            }
        }
    }
}

extern "C" void kernel_launch(void* const* d_in, const int* in_sizes, int n_in,
                              void* d_out, int out_size, void* d_ws, size_t ws_size,
                              hipStream_t stream) {
    const float* h        = (const float*)d_in[0];
    const float* W_left   = (const float*)d_in[1];
    const float* W_right  = (const float*)d_in[2];
    const float* W_top    = (const float*)d_in[3];
    const float* b_conv   = (const float*)d_in[4];
    // d_in[5] = edge_src (implicitly arange(1,N), unused)
    const int*   edge_dest = (const int*)d_in[6];
    const int*   child_pos = (const int*)d_in[7];
    const int*   child_cnt = (const int*)d_in[8];
    float* out = (float*)d_out;
    unsigned short* wsW = (unsigned short*)d_ws;   // 384 KB bf16 packed weights

    const int n4 = N_NODES * HSZ / 4;              // 8,388,608 float4
    zero_f4<<<n4 / 256, 256, 0, stream>>>((float4*)out, n4);
    pack_weights<<<(3 * 256 * 256) / 256, 256, 0, stream>>>(W_left, W_right, W_top, wsW);
    gemm_tbcnn<<<N_NODES / MB, 512, SMEM_BYTES, stream>>>(h, (const uint4*)wsW, b_conv,
                                                          edge_dest, child_pos, child_cnt, out);
    relu_f4<<<n4 / 256, 256, 0, stream>>>((float4*)out, n4);
}
